// _ScaledDotProductAttention_8804682957209
// MI455X (gfx1250) — compile-verified, matrix operand order checked by reading, not run on hardware
//
#include <hip/hip_runtime.h>

// Scaled dot-product attention, returns (output, attn_weights).
// bs=4, heads=16, seq=2048, head_dim=64, fp32 in/out.
// Strategy: bf16 WMMA (v_wmma_f32_16x16x32_bf16) with fp32 accum,
// two-pass softmax (online stats, then recompute+write), LDS-staged
// operands pre-swizzled into WMMA per-lane layouts.

#define BH      128     // bs * heads
#define SEQ     2048
#define DHEAD   64
#define QSCALE  0.125f  // 1/sqrt(64), exact power of two -> fold into Q

typedef __attribute__((ext_vector_type(16))) __bf16        v16bf;
typedef __attribute__((ext_vector_type(8)))  float         v8f;
typedef __attribute__((ext_vector_type(8)))  unsigned int  v8u;

// LDS row paddings (in u32 units) chosen for conflict-free ds_load_b128
// while keeping 16B alignment (all row strides are multiples of 4 u32).
#define KROW 36   // kst: 128 n-rows x 32 d-pairs   (18.4 KB)
#define VROW 68   // vst: 64 d-cols x 64 n-pairs    (17.4 KB)
#define WROW 20   // wst: per-wave 16 m-rows x 16 n-pairs (10 KB total)

__device__ __forceinline__ unsigned int pack_bf16(float a, float b) {
  // round-to-nearest-even fp32 -> bf16, packed pair
  unsigned int ua = __float_as_uint(a);
  unsigned int ub = __float_as_uint(b);
  ua += 0x7fffu + ((ua >> 16) & 1u);
  ub += 0x7fffu + ((ub >> 16) & 1u);
  return (ua >> 16) | (ub & 0xffff0000u);
}

__device__ __forceinline__ v8f wmma_bf16(v8u a, v8u b, v8f c) {
  return __builtin_amdgcn_wmma_f32_16x16x32_bf16(
      false, __builtin_bit_cast(v16bf, a),
      false, __builtin_bit_cast(v16bf, b),
      (short)0, c, false, false);
}

__device__ __forceinline__ v8u ld8(const unsigned int* p) {
  // two aligned b128 loads: u32s [0..3] and [8..11] relative to p
  const uint4* p4 = (const uint4*)p;
  uint4 lo = p4[0];
  uint4 hi = p4[2];
  v8u r = {lo.x, lo.y, lo.z, lo.w, hi.x, hi.y, hi.z, hi.w};
  return r;
}

// B operand (K^T tile) for QK: tile t (16 keys), K-half h (d 0-31 / 32-63)
__device__ __forceinline__ v8u load_b_k(const unsigned int* kst, int t, int h,
                                        int half, int l16) {
  int base = (t * 16 + l16) * KROW + h * 16 + half * 4;
  return ld8(kst + base);
}

// B operand (V tile) for PV: 32-key group q32 (0..3), output-col tile c (0..3)
__device__ __forceinline__ v8u load_b_v(const unsigned int* vst, int q32, int c,
                                        int half, int l16) {
  int base = (c * 16 + l16) * VROW + q32 * 16 + half * 4;
  return ld8(vst + base);
}

__global__ __launch_bounds__(256)
void attn_fused_kernel(const float* __restrict__ q,
                       const float* __restrict__ k,
                       const float* __restrict__ v,
                       float* __restrict__ outO,
                       float* __restrict__ outW) {
  __shared__ unsigned int kst[128 * KROW];
  __shared__ unsigned int vst[64 * VROW];
  __shared__ unsigned int wst[8][16 * WROW];

  const int tid  = threadIdx.x;
  const int lane = tid & 31;
  const int wave = tid >> 5;
  const int half = lane >> 4;   // selects lane-half in WMMA layouts
  const int l16  = lane & 15;

  const int bh = blockIdx.y;                       // 0..127
  const int m0 = blockIdx.x * 128 + wave * 16;     // this wave's 16 q-rows

  const float* qbh = q + (size_t)bh * SEQ * DHEAD;
  const float* kbh = k + (size_t)bh * DHEAD * SEQ;   // stored d x s
  const float* vbh = v + (size_t)bh * SEQ * DHEAD;
  float* obh = outO + (size_t)bh * SEQ * DHEAD;
  float* wbh = outW + (size_t)bh * SEQ * SEQ;

  // ---- Q strip (16 rows x 64) -> two bf16 A operands, scale folded (exact)
  v8u qa[2];
  {
    const int qrow = m0 + l16;
    #pragma unroll
    for (int h = 0; h < 2; ++h) {
      #pragma unroll
      for (int vv = 0; vv < 8; ++vv) {
        // A 16x32 bf16 layout: pair index = half*4 + (vv&3) + (vv>=4)*8
        int dpair = h * 16 + half * 4 + ((vv >> 2) << 3) + (vv & 3);
        const float* qp = qbh + (size_t)qrow * DHEAD + dpair * 2;
        qa[h][vv] = pack_bf16(qp[0] * QSCALE, qp[1] * QSCALE);
      }
    }
  }

  // ================= Pass A: online row max / sum(exp) =================
  float mr[8], lr[8];
  #pragma unroll
  for (int i = 0; i < 8; ++i) { mr[i] = -3.0e38f; lr[i] = 0.0f; }

  for (int chunk = 0; chunk < 16; ++chunk) {
    const int n0 = chunk * 128;
    // cooperative K chunk -> LDS, bf16 pairs along d, pre-swizzled B layout
    #pragma unroll
    for (int i = 0; i < 16; ++i) {
      int s  = i * 256 + tid;        // 4096 u32 slots
      int nr = s & 127;              // key within chunk (coalesced over lanes)
      int dp = s >> 7;               // d-pair 0..31
      const float* kp = kbh + (size_t)(dp * 2) * SEQ + n0 + nr;
      kst[nr * KROW + dp] = pack_bf16(kp[0], kp[SEQ]);
    }
    __syncthreads();
    #pragma unroll
    for (int t = 0; t < 8; ++t) {
      v8f s = {};
      s = wmma_bf16(qa[0], load_b_k(kst, t, 0, half, l16), s);
      s = wmma_bf16(qa[1], load_b_k(kst, t, 1, half, l16), s);
      #pragma unroll
      for (int vv = 0; vv < 8; ++vv) {
        float x  = s[vv];
        float mn = fmaxf(mr[vv], x);
        lr[vv] = lr[vv] * __expf(mr[vv] - mn) + __expf(x - mn);
        mr[vv] = mn;
      }
    }
    __syncthreads();
  }

  // butterfly log-sum-exp combine across the 16 lanes of each half
  #pragma unroll
  for (int d = 1; d <= 8; d <<= 1) {
    #pragma unroll
    for (int vv = 0; vv < 8; ++vv) {
      float mo = __shfl_xor(mr[vv], d, 32);
      float lo = __shfl_xor(lr[vv], d, 32);
      float mn = fmaxf(mr[vv], mo);
      lr[vv] = lr[vv] * __expf(mr[vv] - mn) + lo * __expf(mo - mn);
      mr[vv] = mn;
    }
  }
  float rinv[8];
  #pragma unroll
  for (int vv = 0; vv < 8; ++vv) rinv[vv] = 1.0f / lr[vv];

  // ================= Pass B: recompute, write W, accumulate O ==========
  v8f o0 = {}, o1 = {}, o2 = {}, o3 = {};

  for (int chunk = 0; chunk < 16; ++chunk) {
    const int n0 = chunk * 128;
    #pragma unroll
    for (int i = 0; i < 16; ++i) {   // refill K stage
      int s  = i * 256 + tid;
      int nr = s & 127;
      int dp = s >> 7;
      const float* kp = kbh + (size_t)(dp * 2) * SEQ + n0 + nr;
      kst[nr * KROW + dp] = pack_bf16(kp[0], kp[SEQ]);
    }
    #pragma unroll
    for (int i = 0; i < 16; ++i) {   // V chunk -> LDS, pairs along seq
      int s  = i * 256 + tid;
      int dc = s & 63;               // d col (coalesced over lanes)
      int np = s >> 6;               // seq-pair 0..63
      const float* vp = vbh + (size_t)(n0 + 2 * np) * DHEAD + dc;
      vst[dc * VROW + np] = pack_bf16(vp[0], vp[DHEAD]);
    }
    __syncthreads();

    unsigned short* wrow = (unsigned short*)&wst[wave][0];
    #pragma unroll
    for (int t = 0; t < 8; ++t) {
      v8f s = {};
      s = wmma_bf16(qa[0], load_b_k(kst, t, 0, half, l16), s);
      s = wmma_bf16(qa[1], load_b_k(kst, t, 1, half, l16), s);
      const int cl = (t & 1) * 16 + l16;   // column within 32-key group
      #pragma unroll
      for (int vv = 0; vv < 8; ++vv) {
        float w = __expf(s[vv] - mr[vv]) * rinv[vv];
        const int mrow = vv + 8 * half;
        // normalized fp32 weight -> global (rows of 16 contiguous floats)
        wbh[(size_t)(m0 + mrow) * SEQ + n0 + t * 16 + l16] = w;
        // bf16 -> per-wave A-operand staging in LDS
        unsigned int ub = __float_as_uint(w);
        ub += 0x7fffu + ((ub >> 16) & 1u);
        wrow[(mrow * WROW + (cl >> 1)) * 2 + (cl & 1)] = (unsigned short)(ub >> 16);
      }
      if (t & 1) {                         // 32 keys staged -> PV WMMAs
        int q32 = t >> 1;
        v8u aw = ld8(&wst[wave][l16 * WROW + half * 4]);
        o0 = wmma_bf16(aw, load_b_v(vst, q32, 0, half, l16), o0);
        o1 = wmma_bf16(aw, load_b_v(vst, q32, 1, half, l16), o1);
        o2 = wmma_bf16(aw, load_b_v(vst, q32, 2, half, l16), o2);
        o3 = wmma_bf16(aw, load_b_v(vst, q32, 3, half, l16), o3);
      }
    }
    __syncthreads();
  }

  // ---- write O tile (16 x 64), C/D layout: row = vv + 8*half, col = l16
  #pragma unroll
  for (int vv = 0; vv < 8; ++vv) {
    size_t r = (size_t)(m0 + vv + 8 * half) * DHEAD;
    obh[r +  0 + l16] = o0[vv];
    obh[r + 16 + l16] = o1[vv];
    obh[r + 32 + l16] = o2[vv];
    obh[r + 48 + l16] = o3[vv];
  }
}

extern "C" void kernel_launch(void* const* d_in, const int* in_sizes, int n_in,
                              void* d_out, int out_size, void* d_ws, size_t ws_size,
                              hipStream_t stream) {
  (void)in_sizes; (void)n_in; (void)out_size; (void)d_ws; (void)ws_size;
  const float* q = (const float*)d_in[0];
  const float* k = (const float*)d_in[1];
  const float* v = (const float*)d_in[2];
  float* outO = (float*)d_out;                           // [4,16,2048,64]
  float* outW = outO + (size_t)BH * SEQ * DHEAD;         // [4,16,2048,2048]
  dim3 grid(16, BH);     // x: 128-row m-blocks, y: batch*heads
  attn_fused_kernel<<<grid, 256, 0, stream>>>(q, k, v, outO, outW);
}
